// TransformerEmbed_78520592106175
// MI455X (gfx1250) — matrix operand order repaired and checked by reading, not compiled, not run on hardware
//
#include <hip/hip_runtime.h>

// ---------------------------------------------------------------------------
// CDNA5 (gfx1250) implementation of the TransformerEmbed reference.
// wave32, WMMA f32 = f16 x f16 + f32 (v_wmma_f32_16x16x32_f16).
// NN GEMM stages V tiles via LDS using async global->LDS copies
// (global_load_async_to_lds_b128 + s_wait_asynccnt), guarded by __has_builtin.
// ---------------------------------------------------------------------------

typedef __attribute__((ext_vector_type(16))) _Float16 v16h;
typedef __attribute__((ext_vector_type(8)))  float    v8f;
typedef __attribute__((ext_vector_type(4)))  int      v4i;

#define TEMP_   30.0f
#define EPS_    1e-5f
#define SCALE_  0.011048543456039806f   // sqrt(1/(512*4*4))
#define LOG1E4_OVER_256 0.0359778920781211f  // ln(10000)/256

#if __has_builtin(__builtin_amdgcn_global_load_async_to_lds_b128) && \
    __has_builtin(__builtin_amdgcn_s_wait_asynccnt)
#define HAS_ASYNC_LDS 1
#else
#define HAS_ASYNC_LDS 0
#endif

union FragH {
    v16h     v;
    _Float16 h[16];
    uint4    q[2];
};

__device__ __forceinline__ _Float16 f2h(float x) { return (_Float16)x; }

// A fragment (16x32, f16): lane row M, K = half*16 + j.
// Two contiguous 16B loads per lane.
__device__ __forceinline__ v16h load_fragA(const _Float16* base, int ld, int row,
                                           int kb, int lh) {
    const _Float16* p = base + (size_t)row * ld + kb + lh * 16;
    FragH f;
    f.q[0] = *reinterpret_cast<const uint4*>(p);
    f.q[1] = *reinterpret_cast<const uint4*>(p + 8);
    return f.v;
}

// B fragment (32x16, f16) for NT gemm (B stored row-major [N][K]):
// lane column n, K = half*16 + j.  One contiguous 32B region per lane.
__device__ __forceinline__ v16h load_fragB_nt(const _Float16* base, int ld, int row,
                                              int kb, int lh) {
    const _Float16* p = base + (size_t)row * ld + kb + lh * 16;
    FragH f;
    f.q[0] = *reinterpret_cast<const uint4*>(p);
    f.q[1] = *reinterpret_cast<const uint4*>(p + 8);
    return f.v;
}

// ---------------------------------------------------------------------------
// GEMM (NT): C[b][m][n] = alpha * sum_k A[b][m][k] * B[b][n][k]  (+ bias[n])
// A,B f16 row-major; C f32.  Block = 4 waves; wave w owns N-tile (by*64+w*16).
// Row indices are clamped for M/N tails (stores guarded).
// ---------------------------------------------------------------------------
__global__ void gemm_nt_wmma(const _Float16* __restrict__ A,
                             const _Float16* __restrict__ Bm,
                             float* __restrict__ C,
                             const float* __restrict__ bias,
                             int M, int N, int K,
                             int lda, int ldb, int ldc,
                             size_t sA, size_t sB, size_t sC,
                             float alpha) {
    const int lane = threadIdx.x & 31;
    const int wave = threadIdx.x >> 5;
    const int lh   = lane >> 4;
    const int l16  = lane & 15;
    const int mb   = blockIdx.x * 16;
    const int nb   = blockIdx.y * 64 + wave * 16;
    const int b    = blockIdx.z;

    const _Float16* Ab = A + (size_t)b * sA;
    const _Float16* Bb = Bm + (size_t)b * sB;
    float*          Cb = C + (size_t)b * sC;

    int arow = mb + l16; if (arow >= M) arow = M - 1;
    int brow = nb + l16; if (brow >= N) brow = N - 1;

    v8f acc = {0.f, 0.f, 0.f, 0.f, 0.f, 0.f, 0.f, 0.f};
    for (int kb = 0; kb < K; kb += 32) {
        __builtin_prefetch(Ab + (size_t)arow * lda + kb + 64, 0, 1);
        v16h af = load_fragA(Ab, lda, arow, kb, lh);
        v16h bf = load_fragB_nt(Bb, ldb, brow, kb, lh);
        acc = __builtin_amdgcn_wmma_f32_16x16x32_f16(false, af, false, bf,
                                                     (short)0, acc, false, false);
    }

    const int col = nb + l16;
    const float bv = bias ? bias[col] : 0.f;
#pragma unroll
    for (int r = 0; r < 8; ++r) {
        int row = mb + r + 8 * lh;
        if (row < M) Cb[(size_t)row * ldc + col] = alpha * acc[r] + bv;
    }
}

// ---------------------------------------------------------------------------
// GEMM (NN): C[b][m][n] = sum_k A[b][m][k] * B[b][k][n]
// A f16 row-major [M][Kpad] (zero-padded beyond Kvalid); B f16 row-major [K][N].
// The 32x64 B tile is staged cooperatively into LDS (async global->LDS when
// available), then fragments are gathered from LDS.
// ---------------------------------------------------------------------------
__global__ void gemm_nn_wmma(const _Float16* __restrict__ A,
                             const _Float16* __restrict__ Bm,
                             float* __restrict__ C,
                             int M, int N, int Kpad, int Kvalid,
                             int lda, int ldb, int ldc,
                             size_t sA, size_t sB, size_t sC) {
    const int tid  = threadIdx.x;
    const int lane = tid & 31;
    const int wave = tid >> 5;
    const int lh   = lane >> 4;
    const int l16  = lane & 15;
    const int mb   = blockIdx.x * 16;
    const int nb0  = blockIdx.y * 64;
    const int b    = blockIdx.z;

    const _Float16* Ab = A + (size_t)b * sA;
    const _Float16* Bb = Bm + (size_t)b * sB;
    float*          Cb = C + (size_t)b * sC;

    int arow = mb + l16; if (arow >= M) arow = M - 1;
    const int ncol_l = wave * 16 + l16;      // column within the 64-wide tile
    const int bcol   = nb0 + ncol_l;         // N is a multiple of 64 here

    __shared__ __align__(16) _Float16 vtile[32 * 64];   // K x N staging tile

    v8f acc = {0.f, 0.f, 0.f, 0.f, 0.f, 0.f, 0.f, 0.f};
    for (int kb = 0; kb < Kpad; kb += 32) {
        __syncthreads();   // previous iteration's reads complete
        // Stage V[kb..kb+31][nb0..nb0+63]: 256 chunks of 8 f16, 2 per thread.
#pragma unroll
        for (int c = 0; c < 2; ++c) {
            int chunk = tid + c * 128;       // 0..255
            int r  = chunk >> 3;             // K row 0..31
            int cs = (chunk & 7) * 8;        // col segment 0,8,..,56
            int kr = kb + r; if (kr >= Kvalid) kr = Kvalid - 1;  // A pad is 0
            const _Float16* gsrc = Bb + (size_t)kr * ldb + nb0 + cs;
            _Float16* ldst = &vtile[r * 64 + cs];
#if HAS_ASYNC_LDS
            __builtin_amdgcn_global_load_async_to_lds_b128(
                (__attribute__((address_space(1))) v4i*)gsrc,
                (__attribute__((address_space(3))) v4i*)ldst, 0, 0);
#else
            *reinterpret_cast<uint4*>(ldst) = *reinterpret_cast<const uint4*>(gsrc);
#endif
        }
#if HAS_ASYNC_LDS
        __builtin_amdgcn_s_wait_asynccnt(0);
#endif
        __syncthreads();

        __builtin_prefetch(Ab + (size_t)arow * lda + kb + 64, 0, 1);
        v16h af = load_fragA(Ab, lda, arow, kb, lh);
        FragH bf;
#pragma unroll
        for (int j = 0; j < 16; ++j)
            bf.h[j] = vtile[(lh * 16 + j) * 64 + ncol_l];
        acc = __builtin_amdgcn_wmma_f32_16x16x32_f16(false, af, false, bf.v,
                                                     (short)0, acc, false, false);
    }
#pragma unroll
    for (int r = 0; r < 8; ++r) {
        int row = mb + r + 8 * lh;
        if (row < M) Cb[(size_t)row * ldc + bcol] = acc[r];
    }
}

// ---------------------------------------------------------------------------
// x = flatten(feat) + 1e-3 * sine_pos, layout [B][L][D], L = nimg*HW.
// Writes f32 (residual/value path) and f16 (GEMM operand path).
// ---------------------------------------------------------------------------
__global__ void build_x_k(const float* __restrict__ feat,
                          float* __restrict__ xf, _Float16* __restrict__ xh,
                          int nimg, int Bn, int Dd, int HWp) {
    size_t i = (size_t)blockIdx.x * blockDim.x + threadIdx.x;
    size_t total = (size_t)nimg * HWp * Bn * Dd;
    if (i >= total) return;
    int dd = (int)(i % Dd);
    size_t t = i / Dd;
    int L = nimg * HWp;
    int l = (int)(t % L);
    int b = (int)(t / L);
    int img = l / HWp, p = l % HWp;
    float f = feat[(((size_t)img * Bn + b) * Dd + dd) * HWp + p];
    int j = (dd < 256) ? dd : dd - 256;
    float invf = __expf(-(float)j * LOG1E4_OVER_256);
    float ang = (float)(p + 1) * invf;
    float pe = (dd < 256) ? __sinf(ang) : __cosf(ang);
    float v = f + 1e-3f * pe;
    xf[i] = v;
    xh[i] = f2h(v);
}

__global__ void f32_to_f16_k(const float* __restrict__ in,
                             _Float16* __restrict__ out, int n) {
    int i = blockIdx.x * blockDim.x + threadIdx.x;
    if (i < n) out[i] = f2h(in[i]);
}

__global__ void zero_k(float* p, int n) {
    int i = blockIdx.x * blockDim.x + threadIdx.x;
    if (i < n) p[i] = 0.f;
}

// L2-normalize each 128-wide row (one wave per row), f32 in -> f16 out.
__global__ void rownorm128_k(const float* __restrict__ Pin,
                             _Float16* __restrict__ Pout, int nrows) {
    int row = blockIdx.x * 8 + (threadIdx.x >> 5);
    int lane = threadIdx.x & 31;
    if (row >= nrows) return;
    const float4 v = *reinterpret_cast<const float4*>(Pin + (size_t)row * 128 + lane * 4);
    float ss = v.x * v.x + v.y * v.y + v.z * v.z + v.w * v.w;
#pragma unroll
    for (int m = 16; m > 0; m >>= 1) ss += __shfl_xor(ss, m, 32);
    float inv = 1.f / fmaxf(sqrtf(ss), 1e-12f);
    _Float16* o = Pout + (size_t)row * 128 + lane * 4;
    o[0] = f2h(v.x * inv); o[1] = f2h(v.y * inv);
    o[2] = f2h(v.z * inv); o[3] = f2h(v.w * inv);
}

// Softmax over each logits row (length Lk, stored with ld >= Lk, ld <= 2048).
// Writes f16 probabilities, zero-padded out to ld (so padded-K GEMMs are exact).
__global__ void softmax_rows_k(const float* __restrict__ S,
                               _Float16* __restrict__ P,
                               int Lq, int Lk, int ld) {
    const int q = blockIdx.x, b = blockIdx.y, tid = threadIdx.x;
    const float* row = S + ((size_t)b * Lq + q) * ld;
    _Float16* out = P + ((size_t)b * Lq + q) * ld;
    float v[8];
    float mx = -3.0e38f;
#pragma unroll
    for (int i = 0; i < 8; ++i) {
        int idx = tid + i * 256;
        v[i] = (idx < Lk) ? row[idx] : -3.0e38f;
        mx = fmaxf(mx, v[i]);
    }
    __shared__ float red[256];
    red[tid] = mx; __syncthreads();
    for (int s = 128; s > 0; s >>= 1) {
        if (tid < s) red[tid] = fmaxf(red[tid], red[tid + s]);
        __syncthreads();
    }
    mx = red[0]; __syncthreads();
    float sum = 0.f;
#pragma unroll
    for (int i = 0; i < 8; ++i) {
        int idx = tid + i * 256;
        float e = (idx < Lk) ? __expf(v[i] - mx) : 0.f;
        v[i] = e; sum += e;
    }
    red[tid] = sum; __syncthreads();
    for (int s = 128; s > 0; s >>= 1) {
        if (tid < s) red[tid] += red[tid + s];
        __syncthreads();
    }
    float inv = 1.f / red[0];
#pragma unroll
    for (int i = 0; i < 8; ++i) {
        int idx = tid + i * 256;
        if (idx < ld) out[idx] = f2h(v[i] * inv);
    }
}

// mvec[b][q] = sum_k probs[b][q][k] * posvec[b][k]   (rank-1 "mask" attention)
__global__ void gemv_pos_k(const _Float16* __restrict__ Pr,
                           const float* __restrict__ posv,
                           float* __restrict__ mvec,
                           int Lq, int Lk, int ld, int Lmem) {
    const int q = blockIdx.x, b = blockIdx.y;
    const _Float16* row = Pr + ((size_t)b * Lq + q) * ld;
    const float* pv = posv + (size_t)b * Lmem;
    float s = 0.f;
    for (int k = threadIdx.x; k < Lk; k += 256) s += (float)row[k] * pv[k];
    __shared__ float red[256];
    red[threadIdx.x] = s; __syncthreads();
    for (int st = 128; st > 0; st >>= 1) {
        if (threadIdx.x < st) red[threadIdx.x] += red[threadIdx.x + st];
        __syncthreads();
    }
    if (threadIdx.x == 0) mvec[(size_t)b * Lq + q] = red[0];
}

// posv[b][ni*HW + p] = label[ni][b][p]
__global__ void build_posvec_k(const float* __restrict__ label,
                               float* __restrict__ posv,
                               int NI, int Bn, int HWp) {
    int i = blockIdx.x * blockDim.x + threadIdx.x;
    int total = NI * Bn * HWp;
    if (i >= total) return;
    int p = i % HWp, t = i / HWp;
    int b = t % Bn, ni = t / Bn;
    posv[((size_t)b * NI + ni) * HWp + p] = label[((size_t)ni * Bn + b) * HWp + p];
}

// Y[r][d] = f16(X[r][d] * rowv[r])   (memory * pos, precomputed value matrix)
__global__ void scale_rows_f16_k(const float* __restrict__ X,
                                 const float* __restrict__ rowv,
                                 _Float16* __restrict__ Y,
                                 size_t rows, int Dd) {
    size_t i = (size_t)blockIdx.x * blockDim.x + threadIdx.x;
    if (i >= rows * (size_t)Dd) return;
    Y[i] = f2h(X[i] * rowv[i / Dd]);
}

// ---------------------------------------------------------------------------
// InstanceL2Norm, group = (b, ni), over HW*D elems:  pass 1 sum-of-squares.
// mode 0: pre = A + B (elementwise).  mode 1: pre = A * rowscale[b*HW + p].
// ---------------------------------------------------------------------------
__global__ void inorm_ss_k(const float* __restrict__ A,
                           const float* __restrict__ Bv,
                           float* __restrict__ ssbuf,
                           int mode, int NI, int HWp, int Dd) {
    const int g = blockIdx.y, b = g / NI, ni = g % NI;
    const size_t L = (size_t)NI * HWp;
    const size_t base = ((size_t)b * L + (size_t)ni * HWp) * Dd;
    const int count = HWp * Dd;
    float s = 0.f;
    for (int i = blockIdx.x * blockDim.x + threadIdx.x; i < count;
         i += gridDim.x * blockDim.x) {
        float a = A[base + i];
        float pre = (mode == 0) ? (a + Bv[base + i])
                                : (a * Bv[(size_t)b * HWp + (i / Dd)]);
        s += pre * pre;
    }
    __shared__ float red[256];
    red[threadIdx.x] = s; __syncthreads();
    for (int st = 128; st > 0; st >>= 1) {
        if (threadIdx.x < st) red[threadIdx.x] += red[threadIdx.x + st];
        __syncthreads();
    }
    if (threadIdx.x == 0) atomicAdd(&ssbuf[g], red[0]);
}

// pass 2: y = pre * SCALE*sqrt(D*H*W/(ss+eps)); optional f32/f16/transposed out.
__global__ void inorm_apply_k(const float* __restrict__ A,
                              const float* __restrict__ Bv,
                              const float* __restrict__ ssbuf,
                              float* __restrict__ outF,
                              _Float16* __restrict__ outH,
                              float* __restrict__ outT,   // [b][d][hw] (NI==1)
                              int mode, int NI, int HWp, int Dd) {
    const int g = blockIdx.y, b = g / NI, ni = g % NI;
    const size_t L = (size_t)NI * HWp;
    const size_t base = ((size_t)b * L + (size_t)ni * HWp) * Dd;
    const int count = HWp * Dd;
    const float scale = SCALE_ * sqrtf((float)count / (ssbuf[g] + EPS_));
    int i = blockIdx.x * blockDim.x + threadIdx.x;
    if (i >= count) return;
    float a = A[base + i];
    float pre = (mode == 0) ? (a + Bv[base + i])
                            : (a * Bv[(size_t)b * HWp + (i / Dd)]);
    float y = pre * scale;
    if (outF) outF[base + i] = y;
    if (outH) outH[base + i] = f2h(y);
    if (outT) {
        int p = i / Dd, dd = i % Dd;
        outT[((size_t)b * Dd + dd) * HWp + p] = y;
    }
}

// ---------------------------------------------------------------------------
static inline int ceilDiv(int a, int b) { return (a + b - 1) / b; }

extern "C" void kernel_launch(void* const* d_in, const int* in_sizes, int n_in,
                              void* d_out, int out_size, void* d_ws, size_t ws_size,
                              hipStream_t stream) {
    (void)in_sizes; (void)n_in; (void)out_size; (void)ws_size;

    const float* train_feat  = (const float*)d_in[0];
    const float* test_feat   = (const float*)d_in[1];
    const float* train_label = (const float*)d_in[2];
    const float* wk_self     = (const float*)d_in[3];
    const float* bk_self     = (const float*)d_in[4];
    const float* wk_cross    = (const float*)d_in[5];
    const float* bk_cross    = (const float*)d_in[6];
    float* out = (float*)d_out;

    constexpr int NI = 3, Bn = 16, Dd = 512, HWp = 484, KD = 128;
    constexpr int Lm = NI * HWp;     // 1452
    constexpr int LmPad = 1472;      // ceil32(1452)
    constexpr int LdPad = 512;       // ceil32(484)

    char* ws = (char*)d_ws;
    size_t off = 0;
    auto alloc = [&](size_t bytes) -> void* {
        void* p = ws + off;
        off += (bytes + 255) & ~(size_t)255;
        return p;
    };

    // ---- persistent region ------------------------------------------------
    _Float16* whs      = (_Float16*)alloc((size_t)KD * Dd * 2);
    _Float16* whc      = (_Float16*)alloc((size_t)KD * Dd * 2);
    float*    mem_f    = (float*)   alloc((size_t)Bn * Lm * Dd * 4);
    _Float16* mem_h    = (_Float16*)alloc((size_t)Bn * Lm * Dd * 2);
    _Float16* vprime_h = (_Float16*)alloc((size_t)Bn * Lm * Dd * 2);
    _Float16* pkmem_h  = (_Float16*)alloc((size_t)Bn * Lm * KD * 2);
    float*    posv     = (float*)   alloc((size_t)Bn * Lm * 4);
    float*    ssb      = (float*)   alloc(64 * 4);
    const size_t scratch0 = off;

    // ---- encoder scratch --------------------------------------------------
    float*    xe_f  = (float*)   alloc((size_t)Bn * Lm * Dd * 4);
    _Float16* xe_h  = (_Float16*)alloc((size_t)Bn * Lm * Dd * 2);
    float*    Pe_f  = (float*)   alloc((size_t)Bn * Lm * KD * 4);
    _Float16* Pe_h  = (_Float16*)alloc((size_t)Bn * Lm * KD * 2);
    float*    Se    = (float*)   alloc((size_t)Bn * Lm * LmPad * 4);
    _Float16* Pr_e  = (_Float16*)alloc((size_t)Bn * Lm * LmPad * 2);
    float*    atn_e = (float*)   alloc((size_t)Bn * Lm * Dd * 4);

    // ---- decoder scratch (overlays encoder scratch; ordering is stream) ---
    off = scratch0;
    float*    xd_f  = (float*)   alloc((size_t)Bn * HWp * Dd * 4);
    _Float16* xd_h  = (_Float16*)alloc((size_t)Bn * HWp * Dd * 2);
    float*    Pd_f  = (float*)   alloc((size_t)Bn * HWp * KD * 4);
    _Float16* Pd_h  = (_Float16*)alloc((size_t)Bn * HWp * KD * 2);
    float*    Sd    = (float*)   alloc((size_t)Bn * HWp * LmPad * 4);
    _Float16* Prd   = (_Float16*)alloc((size_t)Bn * HWp * LmPad * 2);
    float*    atn_d = (float*)   alloc((size_t)Bn * HWp * Dd * 4);
    float*    x1_f  = (float*)   alloc((size_t)Bn * HWp * Dd * 4);
    _Float16* x1_h  = (_Float16*)alloc((size_t)Bn * HWp * Dd * 2);
    float*    t3b   = (float*)   alloc((size_t)Bn * HWp * Dd * 4);
    float*    t2b   = (float*)   alloc((size_t)Bn * HWp * Dd * 4);
    float*    t4b   = (float*)   alloc((size_t)Bn * HWp * Dd * 4);
    float*    mvec  = (float*)   alloc((size_t)Bn * HWp * 4);

    // ---- weights to f16 ---------------------------------------------------
    f32_to_f16_k<<<ceilDiv(KD * Dd, 256), 256, 0, stream>>>(wk_self,  whs, KD * Dd);
    f32_to_f16_k<<<ceilDiv(KD * Dd, 256), 256, 0, stream>>>(wk_cross, whc, KD * Dd);

    // ======================= Encoder =======================
    build_x_k<<<ceilDiv(Bn * Lm * Dd, 256), 256, 0, stream>>>(
        train_feat, xe_f, xe_h, NI, Bn, Dd, HWp);
    // P = x @ Wk_self^T + bk
    gemm_nt_wmma<<<dim3(ceilDiv(Lm, 16), ceilDiv(KD, 64), Bn), 128, 0, stream>>>(
        xe_h, whs, Pe_f, bk_self, Lm, KD, Dd, Dd, Dd, KD,
        (size_t)Lm * Dd, 0, (size_t)Lm * KD, 1.f);
    rownorm128_k<<<ceilDiv(Bn * Lm, 8), 256, 0, stream>>>(Pe_f, Pe_h, Bn * Lm);
    // S = TEMP * P @ P^T
    gemm_nt_wmma<<<dim3(ceilDiv(Lm, 16), LmPad / 64, Bn), 128, 0, stream>>>(
        Pe_h, Pe_h, Se, nullptr, Lm, Lm, KD, KD, KD, LmPad,
        (size_t)Lm * KD, (size_t)Lm * KD, (size_t)Lm * LmPad, TEMP_);
    softmax_rows_k<<<dim3(Lm, Bn), 256, 0, stream>>>(Se, Pr_e, Lm, Lm, LmPad);
    // attn = probs @ x
    gemm_nn_wmma<<<dim3(ceilDiv(Lm, 16), Dd / 64, Bn), 128, 0, stream>>>(
        Pr_e, xe_h, atn_e, Lm, Dd, LmPad, Lm, LmPad, Dd, Dd,
        (size_t)Lm * LmPad, (size_t)Lm * Dd, (size_t)Lm * Dd);
    // memory = inorm(x + attn)
    zero_k<<<1, 64, 0, stream>>>(ssb, NI * Bn);
    inorm_ss_k<<<dim3(96, NI * Bn), 256, 0, stream>>>(xe_f, atn_e, ssb, 0, NI, HWp, Dd);
    inorm_apply_k<<<dim3(ceilDiv(HWp * Dd, 256), NI * Bn), 256, 0, stream>>>(
        xe_f, atn_e, ssb, mem_f, mem_h, nullptr, 0, NI, HWp, Dd);
    // pos vector + V' = memory * pos (loop-invariant across decoders)
    build_posvec_k<<<ceilDiv(NI * Bn * HWp, 256), 256, 0, stream>>>(
        train_label, posv, NI, Bn, HWp);
    scale_rows_f16_k<<<ceilDiv(Bn * Lm * Dd, 256), 256, 0, stream>>>(
        mem_f, posv, vprime_h, (size_t)Bn * Lm, Dd);
    // Pk_mem = normalize(memory @ Wk_cross^T + bk_cross)  (loop-invariant)
    gemm_nt_wmma<<<dim3(ceilDiv(Lm, 16), ceilDiv(KD, 64), Bn), 128, 0, stream>>>(
        mem_h, whc, Pe_f, bk_cross, Lm, KD, Dd, Dd, Dd, KD,
        (size_t)Lm * Dd, 0, (size_t)Lm * KD, 1.f);
    rownorm128_k<<<ceilDiv(Bn * Lm, 8), 256, 0, stream>>>(Pe_f, pkmem_h, Bn * Lm);

    // ======================= Decoders =======================
    auto run_decoder = [&](const float* feat, int slot) {
        build_x_k<<<ceilDiv(Bn * HWp * Dd, 256), 256, 0, stream>>>(
            feat, xd_f, xd_h, 1, Bn, Dd, HWp);
        // self-attention
        gemm_nt_wmma<<<dim3(ceilDiv(HWp, 16), ceilDiv(KD, 64), Bn), 128, 0, stream>>>(
            xd_h, whs, Pd_f, bk_self, HWp, KD, Dd, Dd, Dd, KD,
            (size_t)HWp * Dd, 0, (size_t)HWp * KD, 1.f);
        rownorm128_k<<<ceilDiv(Bn * HWp, 8), 256, 0, stream>>>(Pd_f, Pd_h, Bn * HWp);
        gemm_nt_wmma<<<dim3(ceilDiv(HWp, 16), LdPad / 64, Bn), 128, 0, stream>>>(
            Pd_h, Pd_h, Sd, nullptr, HWp, HWp, KD, KD, KD, LdPad,
            (size_t)HWp * KD, (size_t)HWp * KD, (size_t)HWp * LdPad, TEMP_);
        softmax_rows_k<<<dim3(HWp, Bn), 256, 0, stream>>>(Sd, Prd, HWp, HWp, LdPad);
        gemm_nn_wmma<<<dim3(ceilDiv(HWp, 16), Dd / 64, Bn), 128, 0, stream>>>(
            Prd, xd_h, atn_d, HWp, Dd, LdPad, HWp, LdPad, Dd, Dd,
            (size_t)HWp * LdPad, (size_t)HWp * Dd, (size_t)HWp * Dd);
        // x1 = inorm(x + attn)
        zero_k<<<1, 64, 0, stream>>>(ssb, Bn);
        inorm_ss_k<<<dim3(96, Bn), 256, 0, stream>>>(xd_f, atn_d, ssb, 0, 1, HWp, Dd);
        inorm_apply_k<<<dim3(ceilDiv(HWp * Dd, 256), Bn), 256, 0, stream>>>(
            xd_f, atn_d, ssb, x1_f, x1_h, nullptr, 0, 1, HWp, Dd);
        // cross attention (logits shared by mask & t3)
        gemm_nt_wmma<<<dim3(ceilDiv(HWp, 16), ceilDiv(KD, 64), Bn), 128, 0, stream>>>(
            x1_h, whc, Pd_f, bk_cross, HWp, KD, Dd, Dd, Dd, KD,
            (size_t)HWp * Dd, 0, (size_t)HWp * KD, 1.f);
        rownorm128_k<<<ceilDiv(Bn * HWp, 8), 256, 0, stream>>>(Pd_f, Pd_h, Bn * HWp);
        gemm_nt_wmma<<<dim3(ceilDiv(HWp, 16), LmPad / 64, Bn), 128, 0, stream>>>(
            Pd_h, pkmem_h, Sd, nullptr, HWp, Lm, KD, KD, KD, LmPad,
            (size_t)HWp * KD, (size_t)Lm * KD, (size_t)HWp * LmPad, TEMP_);
        softmax_rows_k<<<dim3(HWp, Bn), 256, 0, stream>>>(Sd, Prd, HWp, Lm, LmPad);
        // mask is rank-1: mvec = probs @ posvec
        gemv_pos_k<<<dim3(HWp, Bn), 256, 0, stream>>>(Prd, posv, mvec, HWp, Lm, LmPad, Lm);
        // t3 = probs @ (memory * pos)
        gemm_nn_wmma<<<dim3(ceilDiv(HWp, 16), Dd / 64, Bn), 128, 0, stream>>>(
            Prd, vprime_h, t3b, HWp, Dd, LmPad, Lm, LmPad, Dd, Dd,
            (size_t)HWp * LmPad, (size_t)Lm * Dd, (size_t)HWp * Dd);
        // t2 = inorm(x1 * mvec)
        zero_k<<<1, 64, 0, stream>>>(ssb, Bn);
        inorm_ss_k<<<dim3(96, Bn), 256, 0, stream>>>(x1_f, mvec, ssb, 1, 1, HWp, Dd);
        inorm_apply_k<<<dim3(ceilDiv(HWp * Dd, 256), Bn), 256, 0, stream>>>(
            x1_f, mvec, ssb, t2b, nullptr, nullptr, 1, 1, HWp, Dd);
        // t4 = inorm(x1 + t3)
        zero_k<<<1, 64, 0, stream>>>(ssb, Bn);
        inorm_ss_k<<<dim3(96, Bn), 256, 0, stream>>>(x1_f, t3b, ssb, 0, 1, HWp, Dd);
        inorm_apply_k<<<dim3(ceilDiv(HWp * Dd, 256), Bn), 256, 0, stream>>>(
            x1_f, t3b, ssb, t4b, nullptr, nullptr, 0, 1, HWp, Dd);
        // out = inorm(t2 + t4), transposed to [b][d][hw]
        zero_k<<<1, 64, 0, stream>>>(ssb, Bn);
        inorm_ss_k<<<dim3(96, Bn), 256, 0, stream>>>(t2b, t4b, ssb, 0, 1, HWp, Dd);
        inorm_apply_k<<<dim3(ceilDiv(HWp * Dd, 256), Bn), 256, 0, stream>>>(
            t2b, t4b, ssb, nullptr, nullptr,
            out + (size_t)slot * Bn * Dd * HWp, 0, 1, HWp, Dd);
    };

    for (int img = 0; img < NI; ++img)
        run_decoder(train_feat + (size_t)img * Bn * Dd * HWp, img);
    for (int img = 0; img < NI; ++img)
        run_decoder(test_feat + (size_t)img * Bn * Dd * HWp, NI + img);
}
